// SMoE_31937376813283
// MI455X (gfx1250) — compile-verified, matrix operand order checked by reading, not run on hardware
//
#include <hip/hip_runtime.h>
#include <math.h>

// ---------- constants ----------
constexpr int kD   = 512;    // model dim
constexpr int kE   = 16;     // experts
constexpr int k2D  = 1024;   // hidden (2*D)
constexpr int kMT  = 64;     // tokens per FFN tile (4 M-blocks of 16)

// ---------- types ----------
typedef __bf16 v8bf  __attribute__((ext_vector_type(8)));
typedef __bf16 v16bf __attribute__((ext_vector_type(16)));
typedef float  v8f   __attribute__((ext_vector_type(8)));

// ---------- helpers ----------
__device__ inline v8f wmma_bf16(v16bf a, v16bf b, v8f c) {
  // (neg_a, A, neg_b, B, c_mod, C, reuse_a, reuse_b)
  return __builtin_amdgcn_wmma_f32_16x16x32_bf16(false, a, false, b, (short)0, c,
                                                 false, false);
}

__device__ inline unsigned short f2bf(float f) {  // round-to-nearest-even
  unsigned u = __float_as_uint(f);
  u += 0x7fffu + ((u >> 16) & 1u);
  return (unsigned short)(u >> 16);
}

// Load one bf16 fragment slice: 8 elems at p, 8 elems at p+16 (elements).
// Matches ISA layout: lanes 0-15 hold K 0-7 & 16-23, lanes 16-31 hold K 8-15 & 24-31
// when 'p' already includes the per-lane (lhalf*8) offset.
__device__ inline v16bf load_frag16(const unsigned short* p) {
  v8bf lo = *(const v8bf*)(p);
  v8bf hi = *(const v8bf*)(p + 16);
  return __builtin_shufflevector(lo, hi, 0, 1, 2, 3, 4, 5, 6, 7,
                                 8, 9, 10, 11, 12, 13, 14, 15);
}

// ---------- kernel 0: zero output + counters ----------
__global__ void k_zero(float* __restrict__ out, size_t n, int* __restrict__ counts) {
  size_t i = (size_t)blockIdx.x * blockDim.x + threadIdx.x;
  size_t stride = (size_t)gridDim.x * blockDim.x;
  for (; i < n; i += stride) out[i] = 0.0f;
  if (blockIdx.x == 0 && threadIdx.x < kE) counts[threadIdx.x] = 0;
}

// ---------- kernel 1: f32 -> bf16 (RNE) ----------
__global__ void k_f32_to_bf16(const float* __restrict__ src,
                              unsigned short* __restrict__ dst, size_t n) {
  size_t i = (size_t)blockIdx.x * blockDim.x + threadIdx.x;
  size_t stride = (size_t)gridDim.x * blockDim.x;
  for (; i < n; i += stride) dst[i] = f2bf(src[i]);
}

// ---------- kernel 2: router (wave32 per token) ----------
__global__ __launch_bounds__(256) void k_router(
    const float* __restrict__ x, const float* __restrict__ Wr,
    const float* __restrict__ br, int N, int* __restrict__ counts,
    int* __restrict__ btok, float* __restrict__ bgate,
    float* __restrict__ zloss) {
  int wave = threadIdx.x >> 5;
  int lane = threadIdx.x & 31;
  int t = blockIdx.x * 8 + wave;
  if (t >= N) return;

  const float* xr = x + (size_t)t * kD;
  float acc[kE];
#pragma unroll
  for (int e = 0; e < kE; ++e) acc[e] = 0.0f;

  for (int i = 0; i < kD / 32; ++i) {
    int d = lane + i * 32;
    float xv = xr[d];
#pragma unroll
    for (int e = 0; e < kE; ++e) acc[e] += xv * Wr[e * kD + d];
  }
#pragma unroll
  for (int e = 0; e < kE; ++e) {
#pragma unroll
    for (int off = 16; off > 0; off >>= 1) acc[e] += __shfl_xor(acc[e], off, 32);
  }

  if (lane == 0) {
    float lg[kE];
#pragma unroll
    for (int e = 0; e < kE; ++e) lg[e] = acc[e] + br[e];
    // top-2 with first-occurrence tie break (matches lax.top_k)
    int i1 = 0; float v1 = lg[0];
    for (int e = 1; e < kE; ++e) if (lg[e] > v1) { v1 = lg[e]; i1 = e; }
    int i2 = (i1 == 0) ? 1 : 0; float v2 = lg[i2];
    for (int e = 0; e < kE; ++e)
      if (e != i1 && lg[e] > v2) { v2 = lg[e]; i2 = e; }

    float r  = expf(v2 - v1);           // exp(NEG terms) underflow to 0 exactly
    float g1 = 1.0f / (1.0f + r);
    float g2 = r * g1;
    float z  = v1 + log1pf(r);          // logsumexp over sparse row
    atomicAdd(zloss, z * z / (float)N);

    int s1 = atomicAdd(&counts[i1], 1);
    btok[(size_t)i1 * N + s1] = t; bgate[(size_t)i1 * N + s1] = g1;
    int s2 = atomicAdd(&counts[i2], 1);
    btok[(size_t)i2 * N + s2] = t; bgate[(size_t)i2 * N + s2] = g2;
  }
}

// ---------- kernel 3: expert FFN (gather -> GLU -> project -> scatter) ----------
// 8 waves: each wave owns a distinct 64-column group (ng = wave id) and all
// 4 M-blocks of the 64-token tile, so every weight fragment feeds 8 (Phase A)
// or 4 (Phase B) WMMAs and no two waves stream the same weight rows.
__global__ __launch_bounds__(256) void k_ffn(
    const unsigned short* __restrict__ Xb,   // [N, D] bf16
    const unsigned short* __restrict__ W1b,  // [E, 2D, D] bf16
    const unsigned short* __restrict__ W2b,  // [E, D, D] bf16
    const float* __restrict__ bias_e,        // [E, D]
    const float* __restrict__ bout,          // [E, D]
    const int* __restrict__ counts,
    const int* __restrict__ btok, const float* __restrict__ bgate,
    float* __restrict__ out, int N) {
  extern __shared__ char smem[];
  unsigned short* xs = (unsigned short*)smem;   // 64 KB gathered x tile (bf16)
  unsigned short* ys = xs + kMT * kD;           // 64 KB GLU output tile (bf16)

  const int e = blockIdx.y;
  const int ne = counts[e];
  const int base_slot = blockIdx.x * kMT;
  if (base_slot >= ne) return;

  const int tid = threadIdx.x;

  // gather token rows (bf16) into LDS; padded rows replay token 0 (gate=0 later)
  for (int i = tid; i < kMT * (kD / 8); i += 256) {
    int row = i >> 6, seg = i & 63;
    int slot = base_slot + row;
    int tok = (slot < ne) ? btok[(size_t)e * N + slot] : 0;
    *(v8bf*)(xs + row * kD + seg * 8) =
        *(const v8bf*)(Xb + (size_t)tok * kD + seg * 8);
  }
  __syncthreads();

  const int w = tid >> 5, lane = tid & 31;
  const int ng = w;                        // wave-owned 64-column group
  const int lhalf = lane >> 4, lmod = lane & 15;
  const int fragoff = lhalf * 8;           // per-lane K sub-offset within 32-block

  const unsigned short* W1e = W1b + (size_t)e * k2D * kD;
  const unsigned short* W2e = W2b + (size_t)e * kD * kD;

  // ---- Phase A: h = x @ W1^T, y = gelu(h1) * h2 * bias_e ----
  for (int jt = 0; jt < 4; ++jt) {
    const int c = ng * 64 + jt * 16;       // output column block (0..511)
    v8f acc1[4] = {};
    v8f acc2[4] = {};
    const unsigned short* arow  = xs  + lmod * kD + fragoff;
    const unsigned short* b1row = W1e + (size_t)(c + lmod) * kD + fragoff;
    const unsigned short* b2row = W1e + (size_t)(c + 512 + lmod) * kD + fragoff;
#pragma unroll 2
    for (int kb = 0; kb < kD / 32; ++kb) {
      v16bf b1 = load_frag16(b1row + kb * 32);
      v16bf b2 = load_frag16(b2row + kb * 32);
#pragma unroll
      for (int mt = 0; mt < 4; ++mt) {
        v16bf a = load_frag16(arow + mt * 16 * kD + kb * 32);
        acc1[mt] = wmma_bf16(a, b1, acc1[mt]);
        acc2[mt] = wmma_bf16(a, b2, acc2[mt]);
      }
    }
    const float be = bias_e[e * kD + c + lmod];
#pragma unroll
    for (int mt = 0; mt < 4; ++mt) {
#pragma unroll
      for (int r = 0; r < 8; ++r) {
        int m = mt * 16 + r + lhalf * 8;
        float h1 = acc1[mt][r], h2 = acc2[mt][r];
        float gl = 0.5f * h1 * (1.0f + erff(h1 * 0.70710678118654752f));  // exact
        ys[m * kD + c + lmod] = f2bf(gl * h2 * be);
      }
    }
  }
  __syncthreads();

  // ---- Phase B: out = y @ W2^T + bout, scaled by gate, scatter-add ----
  for (int jt = 0; jt < 4; ++jt) {
    const int j = ng * 64 + jt * 16;
    v8f acc[4] = {};
    const unsigned short* arow = ys  + lmod * kD + fragoff;
    const unsigned short* brow = W2e + (size_t)(j + lmod) * kD + fragoff;
#pragma unroll 2
    for (int kb = 0; kb < kD / 32; ++kb) {
      v16bf b = load_frag16(brow + kb * 32);
#pragma unroll
      for (int mt = 0; mt < 4; ++mt) {
        acc[mt] = wmma_bf16(load_frag16(arow + mt * 16 * kD + kb * 32), b, acc[mt]);
      }
    }
    const float b2v = bout[e * kD + j + lmod];
#pragma unroll
    for (int mt = 0; mt < 4; ++mt) {
#pragma unroll
      for (int r = 0; r < 8; ++r) {
        int m = mt * 16 + r + lhalf * 8;
        int slot = base_slot + m;
        int tok = 0; float g = 0.0f;
        if (slot < ne) {
          tok = btok[(size_t)e * N + slot];
          g   = bgate[(size_t)e * N + slot];
        }
        atomicAdd(out + (size_t)tok * kD + j + lmod, (acc[mt][r] + b2v) * g);
      }
    }
  }
}

// ---------- launch ----------
extern "C" void kernel_launch(void* const* d_in, const int* in_sizes, int n_in,
                              void* d_out, int out_size, void* d_ws, size_t ws_size,
                              hipStream_t stream) {
  const float* x      = (const float*)d_in[0];
  const float* Wr     = (const float*)d_in[1];
  const float* br     = (const float*)d_in[2];
  const float* Win    = (const float*)d_in[3];
  const float* bias_e = (const float*)d_in[4];
  const float* Wout   = (const float*)d_in[5];
  const float* bout   = (const float*)d_in[6];
  float* out = (float*)d_out;

  const int N = in_sizes[0] / kD;          // 32768 tokens
  const size_t nX    = (size_t)in_sizes[0];
  const size_t nWin  = (size_t)in_sizes[3];
  const size_t nWout = (size_t)in_sizes[5];

  char* ws = (char*)d_ws;
  unsigned short* W1b = (unsigned short*)ws;  ws += nWin  * 2;
  unsigned short* W2b = (unsigned short*)ws;  ws += nWout * 2;
  unsigned short* Xb  = (unsigned short*)ws;  ws += nX    * 2;
  int*   btok  = (int*)ws;                    ws += (size_t)kE * N * 4;
  float* bgate = (float*)ws;                  ws += (size_t)kE * N * 4;
  int*   counts = (int*)ws;

  // 1) zero output (N*D values + z_loss) and expert counters
  k_zero<<<2048, 256, 0, stream>>>(out, (size_t)N * kD + 1, counts);
  // 2) bf16 operand staging
  k_f32_to_bf16<<<2048, 256, 0, stream>>>(Win,  W1b, nWin);
  k_f32_to_bf16<<<2048, 256, 0, stream>>>(Wout, W2b, nWout);
  k_f32_to_bf16<<<2048, 256, 0, stream>>>(x,    Xb,  nX);
  // 3) router: logits, top-2 gates, z_loss, bucket scatter
  k_router<<<(N + 7) / 8, 256, 0, stream>>>(x, Wr, br, N, counts, btok, bgate,
                                            out + (size_t)N * kD);
  // 4) expert FFN over routed tiles (128 KB dynamic LDS: x-tile + y-tile)
  const size_t lds_bytes = (size_t)kMT * kD * 2 * 2;  // 131072
  k_ffn<<<dim3(N / kMT, kE), 256, lds_bytes, stream>>>(Xb, W1b, W2b, bias_e, bout,
                                                       counts, btok, bgate, out, N);
}
